// NodeEmbedder_62706522521974
// MI455X (gfx1250) — matrix operand order repaired and checked by reading, not compiled, not run on hardware
//
#include <hip/hip_runtime.h>

typedef __attribute__((ext_vector_type(16))) __bf16 v16bf;
typedef __attribute__((ext_vector_type(8)))  float  v8f;

#define HCW   128   // concat width
#define NHEAD 4
#define HIDC  32
#define EDIM  16

__device__ __forceinline__ unsigned short f2bf(float f) {
    unsigned int u = __float_as_uint(f);
    u += 0x7FFFu + ((u >> 16) & 1u);          // round-to-nearest-even
    return (unsigned short)(u >> 16);
}
__device__ __forceinline__ float bf2f(unsigned short h) {
    return __uint_as_float(((unsigned int)h) << 16);
}
__device__ __forceinline__ void atomicMaxF(float* addr, float val) {
    if (val >= 0.0f) atomicMax((int*)addr, __float_as_int(val));
    else             atomicMin((unsigned int*)addr, __float_as_uint(val));
}

// ---------------------------------------------------------------- init
__global__ void init_buffers(float* stats, float* amax, float* denom,
                             float* agg, int n) {
    long long i = (long long)blockIdx.x * blockDim.x + threadIdx.x;
    long long tot = (long long)n * HCW;
    if (i < 256) stats[i] = 0.0f;
    if (i < (long long)n * NHEAD) { amax[i] = -3.0e38f; denom[i] = 0.0f; }
    if (i < tot) agg[i] = 0.0f;
}

// ---------------------------------------------------------------- batchnorm
__global__ void bn_stats(const float* __restrict__ x, float* stats, int n) {
    int col = threadIdx.x & 127;
    int r0  = blockIdx.x * 2 + (threadIdx.x >> 7);
    int stride = gridDim.x * 2;
    float s = 0.0f, s2 = 0.0f;
    for (int r = r0; r < n; r += stride) {
        float v = x[(long long)r * HCW + col];
        s += v; s2 += v * v;
    }
    atomicAdd(&stats[col], s);
    atomicAdd(&stats[HCW + col], s2);
}

__global__ void bn_apply(const float* __restrict__ x, const float* __restrict__ stats,
                         const float* __restrict__ g, const float* __restrict__ b,
                         unsigned short* __restrict__ xb, int n) {
    long long i = (long long)blockIdx.x * blockDim.x + threadIdx.x;
    if (i >= (long long)n * HCW) return;
    int c = (int)(i & 127);
    float inv_n = 1.0f / (float)n;
    float mu  = stats[c] * inv_n;
    float var = stats[HCW + c] * inv_n - mu * mu;
    float y = (x[i] - mu) * rsqrtf(var + 1e-5f) * g[c] + b[c];
    xb[i] = f2bf(y);
}

// ------------------------------------------------- weight pre-pack to B-fragment
// out[((kc*ntiles + nt)*32 + lane)*8 + v] packs W[K,N] with
// K = kc*32 + (lane>>4)*16 + 2v(+1),  N = nt*16 + (lane&15)
__global__ void pack_weights(const float* __restrict__ W, unsigned int* __restrict__ out,
                             int K, int Nc) {
    int tid = blockIdx.x * blockDim.x + threadIdx.x;
    int total = (K * Nc) >> 1;
    if (tid >= total) return;
    int ntiles = Nc >> 4;
    int v    = tid & 7;
    int lane = (tid >> 3) & 31;
    int rem  = tid >> 8;
    int nt = rem % ntiles;
    int kc = rem / ntiles;
    int k = kc * 32 + (lane >> 4) * 16 + v * 2;
    int n = nt * 16 + (lane & 15);
    unsigned int lo = f2bf(W[(long long)k * Nc + n]);
    unsigned int hi = f2bf(W[(long long)(k + 1) * Nc + n]);
    out[tid] = lo | (hi << 16);
}

// ---------------------------------------------------------------- WMMA GEMM
// C[M,Nc] = A[M,K](bf16) x Bpk(pre-packed bf16) + bias, optional relu.
// One wave per (16-row block, 16-col tile). blockDim.x = ntiles*32.
__global__ void wmma_gemm(const unsigned short* __restrict__ A,
                          const unsigned int*  __restrict__ Bpk,
                          const float* __restrict__ bias,
                          float* __restrict__ Cf, unsigned short* __restrict__ Cb,
                          int M, int K, int Nc, int relu) {
    int ntiles = Nc >> 4;
    int nt   = threadIdx.x >> 5;
    int lane = threadIdx.x & 31;
    if (nt >= ntiles) return;
    int rb = blockIdx.x;
    int row = rb * 16 + (lane & 15);
    if (row >= M) row = M - 1;                 // clamp for safe loads
    int kh = lane >> 4;

    union { v8f f; float e[8]; } c;
#pragma unroll
    for (int i = 0; i < 8; ++i) c.e[i] = 0.0f;

    int kchunks = K >> 5;
    for (int kc = 0; kc < kchunks; ++kc) {
        union { v16bf bf; uint4 q[2]; } a, b;
        // A 16x32 bf16 fragment: lane row = lane&15, K-halves select by lane>>4
        const unsigned short* ap = A + (long long)row * K + kc * 32 + kh * 16;
        a.q[0] = *(const uint4*)(ap);          // K = kc*32 + kh*16 + 0..7
        a.q[1] = *(const uint4*)(ap + 8);      // K = kc*32 + kh*16 + 8..15
        // B fragment: fully linear in pre-packed buffer
        const unsigned int* bp = Bpk + (((long long)kc * ntiles + nt) * 32 + lane) * 8;
        b.q[0] = *(const uint4*)(bp);
        b.q[1] = *(const uint4*)(bp + 4);
        c.f = __builtin_amdgcn_wmma_f32_16x16x32_bf16(
                  false, a.bf, false, b.bf, (short)0, c.f, false, false);
    }

    int colg  = nt * 16 + (lane & 15);
    int rbase = rb * 16 + kh * 8;
    float bi = bias ? bias[colg] : 0.0f;
#pragma unroll
    for (int r = 0; r < 8; ++r) {
        int rowg = rbase + r;
        if (rowg >= M) continue;
        float v = c.e[r] + bi;
        if (relu) v = fmaxf(v, 0.0f);
        long long oi = (long long)rowg * Nc + colg;
        if (Cf) Cf[oi] = v;
        if (Cb) Cb[oi] = f2bf(v);
    }
}

// ---------------------------------------------------------------- edge pass A
// One wave per edge: alpha[e,h] = sum_c att[h,c]*lrelu(xl[src]+xr[dst]+ea@We)
__global__ void edge_alpha(const int* __restrict__ ei, const float* __restrict__ ea,
                           const unsigned short* __restrict__ xl,
                           const unsigned short* __restrict__ xr,
                           const float* __restrict__ We, const float* __restrict__ att,
                           float* __restrict__ alpha, float* __restrict__ amax, int E) {
    int e = (int)(((long long)blockIdx.x * blockDim.x + threadIdx.x) >> 5);
    if (e >= E) return;
    int lane = threadIdx.x & 31;
    int src = ei[e], dst = ei[E + e];
    int c0 = lane * 4;
    int h  = lane >> 3;

    float eav[EDIM];
#pragma unroll
    for (int d = 0; d < EDIM; ++d) eav[d] = ea[(long long)e * EDIM + d];

    float s = 0.0f;
#pragma unroll
    for (int j = 0; j < 4; ++j) {
        int c = c0 + j;
        float m = bf2f(xl[(long long)src * HCW + c]) + bf2f(xr[(long long)dst * HCW + c]);
#pragma unroll
        for (int d = 0; d < EDIM; ++d) m += eav[d] * We[d * HCW + c];
        float lr = m > 0.0f ? m : 0.2f * m;
        s += att[h * HIDC + (c & 31)] * lr;
    }
#pragma unroll
    for (int off = 1; off < 8; off <<= 1) s += __shfl_xor(s, off, 32);
    if ((lane & 7) == 0) {
        alpha[(long long)e * NHEAD + h] = s;
        atomicMaxF(&amax[(long long)dst * NHEAD + h], s);
    }
}

// ---------------------------------------------------------------- edge pass B
// agg[dst] += xl[src] * exp(alpha - amax[dst]); denom[dst] += exp(...)
__global__ void edge_msg(const int* __restrict__ ei,
                         const unsigned short* __restrict__ xl,
                         const float* __restrict__ alpha, const float* __restrict__ amax,
                         float* __restrict__ denom, float* __restrict__ agg, int E) {
    int e = (int)(((long long)blockIdx.x * blockDim.x + threadIdx.x) >> 5);
    if (e >= E) return;
    int lane = threadIdx.x & 31;
    int src = ei[e], dst = ei[E + e];
    int h  = lane >> 3;
    int c0 = lane * 4;
    float w = __expf(alpha[(long long)e * NHEAD + h] - amax[(long long)dst * NHEAD + h]);
    if ((lane & 7) == 0) atomicAdd(&denom[(long long)dst * NHEAD + h], w);
#pragma unroll
    for (int j = 0; j < 4; ++j) {
        int c = c0 + j;
        atomicAdd(&agg[(long long)dst * HCW + c],
                  bf2f(xl[(long long)src * HCW + c]) * w);
    }
}

// ---------------------------------------------------------------- finalize
__global__ void finalize_nodes(const float* __restrict__ agg, const float* __restrict__ denom,
                               const float* __restrict__ conv_bias,
                               unsigned short* __restrict__ xb2, int n) {
    long long i = (long long)blockIdx.x * blockDim.x + threadIdx.x;
    if (i >= (long long)n * HCW) return;
    int c = (int)(i & 127);
    long long nd = i >> 7;
    int h = c >> 5;
    float v = agg[i] / (denom[nd * NHEAD + h] + 1e-16f) + conv_bias[c];
    xb2[i] = f2bf(v);
}

// ================================================================ host
extern "C" void kernel_launch(void* const* d_in, const int* in_sizes, int n_in,
                              void* d_out, int out_size, void* d_ws, size_t ws_size,
                              hipStream_t stream) {
    const float* x  = (const float*)d_in[0];
    const int*   ei = (const int*)d_in[1];
    const float* ea = (const float*)d_in[2];
    int Nn = in_sizes[0] / HCW;
    int Ee = in_sizes[2] / EDIM;

    // ---- carve workspace
    char* w = (char*)d_ws;
    auto carve = [&](size_t bytes) -> char* {
        char* p = w; w += (bytes + 255) & ~(size_t)255; return p;
    };
    float* stats = (float*)carve(256 * 4);
    float* amax  = (float*)carve((size_t)Nn * NHEAD * 4);
    float* denom = (float*)carve((size_t)Nn * NHEAD * 4);
    float* alpha = (float*)carve((size_t)Ee * NHEAD * 4);
    float* agg   = (float*)carve((size_t)Nn * HCW * 4);
    unsigned short* xb = (unsigned short*)carve((size_t)Nn * HCW * 2);
    unsigned short* xl = (unsigned short*)carve((size_t)Nn * HCW * 2);
    unsigned short* xr = (unsigned short*)carve((size_t)Nn * HCW * 2);
    float* hf = (float*)carve((size_t)Nn * HCW * 4);
    unsigned short* hb   = (unsigned short*)carve((size_t)Nn * HCW * 2);
    unsigned short* hidb = (unsigned short*)carve((size_t)Nn * HIDC * 2);
    unsigned int* wpkA = (unsigned int*)carve(8192 * 4);
    unsigned int* wpkB = (unsigned int*)carve(8192 * 4);
    unsigned int* wpkL = (unsigned int*)carve(8192 * 4);
    unsigned int* wpk1 = (unsigned int*)carve(2048 * 4);
    unsigned int* wpk2 = (unsigned int*)carve(1024 * 4);

    int ew_blocks = (int)(((long long)Ee * 32 + 255) / 256);
    int nb_elem   = (int)(((long long)Nn * HCW + 255) / 256);
    int row_blk   = (Nn + 15) / 16;

    const float* cur = x;
    for (int l = 0; l < 2; ++l) {
        int base = 3 + l * 11;
        const float* bn_g  = (const float*)d_in[base + 0];
        const float* bn_b  = (const float*)d_in[base + 1];
        const float* Wl    = (const float*)d_in[base + 2];
        const float* bl    = (const float*)d_in[base + 3];
        const float* Wr    = (const float*)d_in[base + 4];
        const float* br    = (const float*)d_in[base + 5];
        const float* We    = (const float*)d_in[base + 6];
        const float* att   = (const float*)d_in[base + 7];
        const float* cbias = (const float*)d_in[base + 8];
        const float* linW  = (const float*)d_in[base + 9];
        const float* linb  = (const float*)d_in[base + 10];

        init_buffers<<<nb_elem, 256, 0, stream>>>(stats, amax, denom, agg, Nn);
        bn_stats<<<512, 256, 0, stream>>>(cur, stats, Nn);
        bn_apply<<<nb_elem, 256, 0, stream>>>(cur, stats, bn_g, bn_b, xb, Nn);

        pack_weights<<<(8192 + 255) / 256, 256, 0, stream>>>(Wl,   wpkA, 128, 128);
        pack_weights<<<(8192 + 255) / 256, 256, 0, stream>>>(Wr,   wpkB, 128, 128);
        pack_weights<<<(8192 + 255) / 256, 256, 0, stream>>>(linW, wpkL, 128, 128);

        wmma_gemm<<<row_blk, 256, 0, stream>>>(xb, wpkA, bl, nullptr, xl, Nn, 128, 128, 0);
        wmma_gemm<<<row_blk, 256, 0, stream>>>(xb, wpkB, br, nullptr, xr, Nn, 128, 128, 0);

        edge_alpha<<<ew_blocks, 256, 0, stream>>>(ei, ea, xl, xr, We, att, alpha, amax, Ee);
        edge_msg<<<ew_blocks, 256, 0, stream>>>(ei, xl, alpha, amax, denom, agg, Ee);

        finalize_nodes<<<nb_elem, 256, 0, stream>>>(agg, denom, cbias, xb, Nn);
        wmma_gemm<<<row_blk, 256, 0, stream>>>(xb, wpkL, linb, hf, hb, Nn, 128, 128, 1);
        cur = hf;
    }

    const float* oW1 = (const float*)d_in[25];
    const float* ob1 = (const float*)d_in[26];
    const float* oW2 = (const float*)d_in[27];
    const float* ob2 = (const float*)d_in[28];

    pack_weights<<<(2048 + 255) / 256, 256, 0, stream>>>(oW1, wpk1, 128, 32);
    wmma_gemm<<<row_blk, 64, 0, stream>>>(hb, wpk1, ob1, nullptr, hidb, Nn, 128, 32, 0);
    pack_weights<<<(1024 + 255) / 256, 256, 0, stream>>>(oW2, wpk2, 32, 64);
    wmma_gemm<<<row_blk, 128, 0, stream>>>(hidb, wpk2, ob2, (float*)d_out, nullptr, Nn, 32, 64, 0);
}
